// Block_9122510537354
// MI455X (gfx1250) — matrix operand order repaired and checked by reading, not compiled, not run on hardware
//
#include <hip/hip_runtime.h>
#include <hip/hip_bf16.h>
#include <math.h>

// ---------------------------------------------------------------------------
// Mamba block forward for MI455X (gfx1250).
// All GEMM operands pre-split to bf16 hi/lo (bf16x3 trick ~ fp32 accuracy).
// GEMM: TDM (tensor_load_to_lds) double-buffered LDS staging ->
//       ds_load_b128 fragments -> v_wmma_f32_16x16x32_bf16.
// ---------------------------------------------------------------------------

#define DM     1024
#define DI     2048
#define DSTATE 16
#define DTR    64
#define BSZ    2
#define LSEQ   1024
#define MROWS  (BSZ * LSEQ)          // 2048
#define XROW   (2 * DI)              // 4096 (xz row stride)
#define XDROW  (DTR + 2 * DSTATE)    // 96   (x_dbl row stride)

typedef __attribute__((ext_vector_type(16))) __bf16        bf16x16;
typedef __attribute__((ext_vector_type(8)))  __bf16        bf16x8;
typedef __attribute__((ext_vector_type(8)))  float         v8f;
typedef __attribute__((ext_vector_type(4)))  unsigned int  v4u;
typedef __attribute__((ext_vector_type(8)))  int           v8i;
typedef __attribute__((ext_vector_type(4)))  int           v4i;

union Frag { bf16x16 v; bf16x8 h[2]; };

static __device__ __forceinline__ float sigmoidf_(float x) {
    return 1.0f / (1.0f + expf(-x));
}

static __device__ __forceinline__ v8f wmma_bf16(bf16x16 a, bf16x16 b, v8f c) {
    return __builtin_amdgcn_wmma_f32_16x16x32_bf16(false, a, false, b,
                                                   (short)0, c, false, false);
}

static __device__ __forceinline__ void wait_tensorcnt0() {
#if defined(__has_builtin)
#if __has_builtin(__builtin_amdgcn_s_wait_tensorcnt)
    __builtin_amdgcn_s_wait_tensorcnt((short)0);
    return;
#endif
#endif
    asm volatile("s_wait_tensorcnt 0x0" ::: "memory");
}

#if defined(__has_builtin)
#if __has_builtin(__builtin_amdgcn_tensor_load_to_lds)
#define HAVE_TDM 1
#endif
#endif

#ifdef HAVE_TDM
// 2D TDM tile load: tile_rows x tile_k elements of size (1<<ds_code) bytes,
// row stride `ld` elements, from gaddr (tile origin) into LDS byte offset.
// avail0/avail1: remaining tensor extent from tile origin (OOB reads -> 0).
static __device__ __forceinline__ void tdm_load_2d(
    const void* gaddr, unsigned lds_off, unsigned ds_code,
    int tile_k, int tile_rows, int ld, int avail0, int avail1)
{
    unsigned long long ga = (unsigned long long)(size_t)gaddr;
    v4u g0;
    g0.x = 1u;                                         // count=1, user desc
    g0.y = lds_off;                                    // lds_addr [63:32]
    g0.z = (unsigned)(ga & 0xffffffffu);               // global_addr[31:0]
    g0.w = (unsigned)((ga >> 32) & 0x1ffffffu)         // global_addr[56:32]
         | (2u << 30);                                 // type=2 ("image")
    unsigned w0 = (ds_code << 16);                     // data_size, mask=0
    unsigned w1 = ((unsigned)avail0 & 0xffffu) << 16;  // tensor_dim0[15:0]
    unsigned w2 = (((unsigned)avail0 >> 16) & 0xffffu)
                | (((unsigned)avail1 & 0xffffu) << 16);
    unsigned w3 = (((unsigned)avail1 >> 16) & 0xffffu)
                | (((unsigned)tile_k & 0xffffu) << 16);   // tile_dim0
    unsigned w4 = ((unsigned)tile_rows & 0xffffu);        // tile_dim1 (dim2=0)
    unsigned w5 = (unsigned)ld;                           // dim0_stride[31:0]
    v8i g1 = {(int)w0, (int)w1, (int)w2, (int)w3, (int)w4, (int)w5, 0, 0};
    v4i z  = {0, 0, 0, 0};
#if __has_include(<hip/amd_detail/amd_gfx1250_TDM.h>)
    v8i z8 = {0, 0, 0, 0, 0, 0, 0, 0};
    __builtin_amdgcn_tensor_load_to_lds(g0, g1, z, z, z8, 0);   // clang-23 form
#else
    __builtin_amdgcn_tensor_load_to_lds(g0, g1, z, z, 0);       // ROCm 7.2 form
#endif
}
#endif

// ---------------------------------------------------------------------------
// Kernel 0: bf16 hi/lo split of an fp32 array (for weights).
// ---------------------------------------------------------------------------
__global__ __launch_bounds__(256) void k_split(
    const float* __restrict__ src, __bf16* __restrict__ hi,
    __bf16* __restrict__ lo, int n)
{
    int i = blockIdx.x * 256 + threadIdx.x;
    if (i < n) {
        float x = src[i];
        __bf16 h = (__bf16)x;
        hi[i] = h;
        lo[i] = (__bf16)(x - (float)h);
    }
}

// ---------------------------------------------------------------------------
// Kernel 1: residual = h + residual ; hn = layernorm(residual)*w + b
// hn emitted directly as bf16 hi/lo.
// ---------------------------------------------------------------------------
__global__ __launch_bounds__(256) void k_add_ln(
    const float* __restrict__ h, const float* __restrict__ res,
    const float* __restrict__ w, const float* __restrict__ bia,
    float* __restrict__ res_out,
    __bf16* __restrict__ hn_h, __bf16* __restrict__ hn_l)
{
    int row = blockIdx.x;
    int tid = threadIdx.x;
    const float* hp = h   + (size_t)row * DM;
    const float* rp = res + (size_t)row * DM;
    float* ro = res_out + (size_t)row * DM;

    float v[4];
    float s = 0.f, s2 = 0.f;
#pragma unroll
    for (int i = 0; i < 4; ++i) {
        int c = tid + i * 256;
        float x = hp[c] + rp[c];
        v[i] = x;
        ro[c] = x;
        s  += x;
        s2 += x * x;
    }
    __shared__ float sh[256];
    __shared__ float sh2[256];
    sh[tid] = s; sh2[tid] = s2;
    __syncthreads();
    for (int o = 128; o > 0; o >>= 1) {
        if (tid < o) { sh[tid] += sh[tid + o]; sh2[tid] += sh2[tid + o]; }
        __syncthreads();
    }
    float mu  = sh[0] * (1.0f / DM);
    float var = sh2[0] * (1.0f / DM) - mu * mu;
    float inv = rsqrtf(var + 1e-5f);
#pragma unroll
    for (int i = 0; i < 4; ++i) {
        int c = tid + i * 256;
        float y = (v[i] - mu) * inv * w[c] + bia[c];
        __bf16 hh = (__bf16)y;
        size_t o = (size_t)row * DM + c;
        hn_h[o] = hh;
        hn_l[o] = (__bf16)(y - (float)hh);
    }
}

// ---------------------------------------------------------------------------
// Kernel 2: GEMM  C[m][n] = sum_k A[m][k] * W[n][k], operands bf16 hi/lo.
// Block: 256 thr = 8 waves (4 M x 2 N); block tile 128(M) x 64(N); K-step 32.
// TDM double-buffered LDS staging; fragments = 2x ds_load_b128 each;
// 12 v_wmma per k-step per wave (Ah*Bh + Al*Bh + Ah*Bl).
// epi==1: C = softplus(C + bias[n]).  Optional bf16 hi/lo C output.
// Requires M % 128 == 0, K % 32 == 0.
// ---------------------------------------------------------------------------
__global__ __launch_bounds__(256) void k_gemm(
    const __bf16* __restrict__ Ah, const __bf16* __restrict__ Al,
    const __bf16* __restrict__ Wh, const __bf16* __restrict__ Wl,
    float* __restrict__ C, int N, int K, int lda, int ldb, int ldc,
    int epi, const float* __restrict__ bias,
    __bf16* __restrict__ Ch, __bf16* __restrict__ Cl)
{
    __shared__ __bf16 sAh[2][128 * 32];   // 8 KB each
    __shared__ __bf16 sAl[2][128 * 32];
    __shared__ __bf16 sBh[2][64 * 32];    // 4 KB each
    __shared__ __bf16 sBl[2][64 * 32];

    int tid  = threadIdx.x;
    int lane = tid & 31;
    int wv   = tid >> 5;
    int wm   = wv & 3;               // 0..3 : wave M position
    int wn   = wv >> 2;              // 0..1 : wave N position
    int m0b  = blockIdx.y * 128;
    int n0b  = blockIdx.x * 64;
    int half = lane >> 4;
    int idx16 = lane & 15;

    v8f acc[2][2] = {};

#ifdef HAVE_TDM
    if (wv == 0) {
        tdm_load_2d(Ah + (size_t)m0b * lda, (unsigned)(size_t)(void*)sAh[0], 1u,
                    32, 128, lda, lda, MROWS - m0b);
        tdm_load_2d(Al + (size_t)m0b * lda, (unsigned)(size_t)(void*)sAl[0], 1u,
                    32, 128, lda, lda, MROWS - m0b);
        tdm_load_2d(Wh + (size_t)n0b * ldb, (unsigned)(size_t)(void*)sBh[0], 1u,
                    32, 64, ldb, K, N - n0b);
        tdm_load_2d(Wl + (size_t)n0b * ldb, (unsigned)(size_t)(void*)sBl[0], 1u,
                    32, 64, ldb, K, N - n0b);
        wait_tensorcnt0();
    }
    __syncthreads();
    int buf = 0;
#endif

    for (int k0 = 0; k0 < K; k0 += 32) {
#ifdef HAVE_TDM
        int nxt = buf ^ 1;
        if (k0 + 32 < K && wv == 0) {           // prefetch next tiles (async)
            int k1 = k0 + 32;
            tdm_load_2d(Ah + (size_t)m0b * lda + k1,
                        (unsigned)(size_t)(void*)sAh[nxt], 1u,
                        32, 128, lda, lda - k1, MROWS - m0b);
            tdm_load_2d(Al + (size_t)m0b * lda + k1,
                        (unsigned)(size_t)(void*)sAl[nxt], 1u,
                        32, 128, lda, lda - k1, MROWS - m0b);
            tdm_load_2d(Wh + (size_t)n0b * ldb + k1,
                        (unsigned)(size_t)(void*)sBh[nxt], 1u,
                        32, 64, ldb, K - k1, N - n0b);
            tdm_load_2d(Wl + (size_t)n0b * ldb + k1,
                        (unsigned)(size_t)(void*)sBl[nxt], 1u,
                        32, 64, ldb, K - k1, N - n0b);
        }
#else
        const int buf = 0;
        __syncthreads();
        for (int i2 = tid; i2 < 128 * 32; i2 += 256) {
            int r = i2 >> 5, c = i2 & 31;
            size_t g = (size_t)(m0b + r) * lda + k0 + c;
            sAh[0][i2] = Ah[g];
            sAl[0][i2] = Al[g];
        }
        for (int i2 = tid; i2 < 64 * 32; i2 += 256) {
            int r = i2 >> 5, c = i2 & 31;
            int nr = n0b + r;
            __bf16 zh = (__bf16)0.f;
            sBh[0][i2] = (nr < N) ? Wh[(size_t)nr * ldb + k0 + c] : zh;
            sBl[0][i2] = (nr < N) ? Wl[(size_t)nr * ldb + k0 + c] : zh;
        }
        __syncthreads();
#endif
        // Fragments from LDS (ISA 7.12.2 layouts -> 2 contiguous 16B reads).
        Frag ah[2], al[2], bh[2], bl[2];
#pragma unroll
        for (int i = 0; i < 2; ++i) {
            const __bf16* rah = &sAh[buf][(wm * 32 + i * 16 + idx16) * 32];
            const __bf16* ral = &sAl[buf][(wm * 32 + i * 16 + idx16) * 32];
            ah[i].h[0] = *(const bf16x8*)(rah + half * 8);
            ah[i].h[1] = *(const bf16x8*)(rah + 16 + half * 8);
            al[i].h[0] = *(const bf16x8*)(ral + half * 8);
            al[i].h[1] = *(const bf16x8*)(ral + 16 + half * 8);
            const __bf16* rbh = &sBh[buf][(wn * 32 + i * 16 + idx16) * 32];
            const __bf16* rbl = &sBl[buf][(wn * 32 + i * 16 + idx16) * 32];
            bh[i].h[0] = *(const bf16x8*)(rbh + half * 8);
            bh[i].h[1] = *(const bf16x8*)(rbh + 16 + half * 8);
            bl[i].h[0] = *(const bf16x8*)(rbl + half * 8);
            bl[i].h[1] = *(const bf16x8*)(rbl + 16 + half * 8);
        }
#pragma unroll
        for (int i = 0; i < 2; ++i)
#pragma unroll
            for (int j = 0; j < 2; ++j) {
                acc[i][j] = wmma_bf16(ah[i].v, bh[j].v, acc[i][j]);
                acc[i][j] = wmma_bf16(al[i].v, bh[j].v, acc[i][j]);
                acc[i][j] = wmma_bf16(ah[i].v, bl[j].v, acc[i][j]);
            }
#ifdef HAVE_TDM
        if (wv == 0) wait_tensorcnt0();
        __syncthreads();
        buf = nxt;
#endif
    }

    // C/D layout: VGPR r -> M = r + half*8, N = idx16
#pragma unroll
    for (int i = 0; i < 2; ++i)
#pragma unroll
        for (int j = 0; j < 2; ++j) {
            int col  = n0b + wn * 32 + j * 16 + idx16;
            int row0 = m0b + wm * 32 + i * 16 + half * 8;
            if (col < N) {
#pragma unroll
                for (int r = 0; r < 8; ++r) {
                    float v = acc[i][j][r];
                    if (epi == 1) {
                        v += bias[col];
                        v = (v > 20.0f) ? v : log1pf(expf(v));   // softplus
                    }
                    size_t o = (size_t)(row0 + r) * ldc + col;
                    C[o] = v;
                    if (Ch) {
                        __bf16 hh = (__bf16)v;
                        Ch[o] = hh;
                        Cl[o] = (__bf16)(v - (float)hh);
                    }
                }
            }
        }
}

// ---------------------------------------------------------------------------
// Kernel 3: causal depthwise conv1d (K=4) + bias + SiLU; f32 + bf16 hi/lo out
// ---------------------------------------------------------------------------
__global__ __launch_bounds__(256) void k_conv_silu(
    const float* __restrict__ xz, const float* __restrict__ cw,
    const float* __restrict__ cb, float* __restrict__ xc,
    __bf16* __restrict__ xch, __bf16* __restrict__ xcl)
{
    int idx = blockIdx.x * 256 + threadIdx.x;          // < BSZ*LSEQ*DI
    int d = idx & (DI - 1);
    int l = (idx >> 11) & (LSEQ - 1);
    int b = idx >> 21;
    float acc = cb[d];
    const float* wr = cw + d * 4;
#pragma unroll
    for (int j = 0; j < 4; ++j) {
        int t = l + j - 3;
        if (t >= 0)
            acc += wr[j] * xz[((size_t)(b * LSEQ + t)) * XROW + d];
    }
    float v = acc * sigmoidf_(acc);                    // SiLU
    xc[idx] = v;
    __bf16 hh = (__bf16)v;
    xch[idx] = hh;
    xcl[idx] = (__bf16)(v - (float)hh);
}

// ---------------------------------------------------------------------------
// Kernel 4: selective scan; y = (y + x*D) * silu(z) -> bf16 hi/lo
// ---------------------------------------------------------------------------
__global__ __launch_bounds__(256) void k_scan(
    const float* __restrict__ dt, const float* __restrict__ xc,
    const float* __restrict__ xdbl, const float* __restrict__ xz,
    const float* __restrict__ A_log, const float* __restrict__ Dw,
    __bf16* __restrict__ yh, __bf16* __restrict__ yl)
{
    int gid = blockIdx.x * 256 + threadIdx.x;          // 0..BSZ*DI-1
    int d = gid & (DI - 1);
    int b = gid >> 11;

    float aA[DSTATE];
#pragma unroll
    for (int s = 0; s < DSTATE; ++s)
        aA[s] = -expf(A_log[d * DSTATE + s]);
    float Dv = Dw[d];

    float st[DSTATE];
#pragma unroll
    for (int s = 0; s < DSTATE; ++s) st[s] = 0.f;

    size_t base = (size_t)b * LSEQ;
    for (int t = 0; t < LSEQ; ++t) {
        size_t r = base + t;
        if (t + 8 < LSEQ) {                            // global_prefetch_b8
            __builtin_prefetch(&dt[(r + 8) * DI + d], 0, 1);
            __builtin_prefetch(&xc[(r + 8) * DI + d], 0, 1);
        }
        float dtv = dt[r * DI + d];
        float xv  = xc[r * DI + d];
        const float* bc = xdbl + r * XDROW;            // B at +64, C at +80
        float y = 0.f;
#pragma unroll
        for (int s = 0; s < DSTATE; ++s) {
            float dA = expf(dtv * aA[s]);
            st[s] = dA * st[s] + (dtv * bc[DTR + s]) * xv;
            y += st[s] * bc[DTR + DSTATE + s];
        }
        float z  = xz[r * XROW + DI + d];
        float ya = (y + xv * Dv) * (z * sigmoidf_(z));
        __bf16 hh = (__bf16)ya;
        yh[r * DI + d] = hh;
        yl[r * DI + d] = (__bf16)(ya - (float)hh);
    }
}

// ---------------------------------------------------------------------------
// Launch
// ---------------------------------------------------------------------------
extern "C" void kernel_launch(void* const* d_in, const int* in_sizes, int n_in,
                              void* d_out, int out_size, void* d_ws, size_t ws_size,
                              hipStream_t stream)
{
    const float* h         = (const float*)d_in[0];
    const float* residual  = (const float*)d_in[1];
    const float* ln_w      = (const float*)d_in[2];
    const float* ln_b      = (const float*)d_in[3];
    const float* in_proj_w = (const float*)d_in[4];   // (4096,1024)
    const float* conv_w    = (const float*)d_in[5];   // (2048,1,4)
    const float* conv_b    = (const float*)d_in[6];
    const float* x_proj_w  = (const float*)d_in[7];   // (96,2048)
    const float* dt_proj_w = (const float*)d_in[8];   // (2048,64)
    const float* dt_proj_b = (const float*)d_in[9];
    const float* A_log     = (const float*)d_in[10];  // (2048,16)
    const float* Dw        = (const float*)d_in[11];
    const float* out_proj_w= (const float*)d_in[12];  // (1024,2048)

    float* out_main = (float*)d_out;                      // (B,L,DM)
    float* res_out  = out_main + (size_t)BSZ * LSEQ * DM; // (B,L,DM)

    // ---- workspace carve-up -------------------------------------------------
    char* wp = (char*)d_ws;
#define TAKE_F32(name, count)  float*  name = (float*)wp;  wp += (size_t)(count) * 4;
#define TAKE_BF16(name, count) __bf16* name = (__bf16*)wp; wp += (size_t)(count) * 2;
    TAKE_F32 (xz,     (size_t)MROWS * XROW)     // 2048*4096
    TAKE_F32 (xc,     (size_t)MROWS * DI)       // 2048*2048
    TAKE_F32 (xdbl,   (size_t)MROWS * XDROW)    // 2048*96
    TAKE_F32 (dtb,    (size_t)MROWS * DI)       // 2048*2048
    TAKE_BF16(hn_h,   (size_t)MROWS * DM)
    TAKE_BF16(hn_l,   (size_t)MROWS * DM)
    TAKE_BF16(xch,    (size_t)MROWS * DI)
    TAKE_BF16(xcl,    (size_t)MROWS * DI)
    TAKE_BF16(xdbl_h, (size_t)MROWS * XDROW)
    TAKE_BF16(xdbl_l, (size_t)MROWS * XDROW)
    TAKE_BF16(yh,     (size_t)MROWS * DI)
    TAKE_BF16(yl,     (size_t)MROWS * DI)
    TAKE_BF16(ipw_h,  (size_t)XROW * DM)        // 4096*1024
    TAKE_BF16(ipw_l,  (size_t)XROW * DM)
    TAKE_BF16(xpw_h,  (size_t)XDROW * DI)       // 96*2048
    TAKE_BF16(xpw_l,  (size_t)XDROW * DI)
    TAKE_BF16(dtw_h,  (size_t)DI * DTR)         // 2048*64
    TAKE_BF16(dtw_l,  (size_t)DI * DTR)
    TAKE_BF16(opw_h,  (size_t)DM * DI)          // 1024*2048
    TAKE_BF16(opw_l,  (size_t)DM * DI)
#undef TAKE_F32
#undef TAKE_BF16

    // 0) split weights to bf16 hi/lo (once per call)
    int n1 = XROW * DM, n2 = XDROW * DI, n3 = DI * DTR, n4 = DM * DI;
    k_split<<<(n1 + 255) / 256, 256, 0, stream>>>(in_proj_w,  ipw_h, ipw_l, n1);
    k_split<<<(n2 + 255) / 256, 256, 0, stream>>>(x_proj_w,   xpw_h, xpw_l, n2);
    k_split<<<(n3 + 255) / 256, 256, 0, stream>>>(dt_proj_w,  dtw_h, dtw_l, n3);
    k_split<<<(n4 + 255) / 256, 256, 0, stream>>>(out_proj_w, opw_h, opw_l, n4);

    // 1) residual add + layernorm (-> bf16 hi/lo)
    k_add_ln<<<MROWS, 256, 0, stream>>>(h, residual, ln_w, ln_b, res_out,
                                        hn_h, hn_l);

    // 2) xz = hn @ in_proj_w^T   (M=2048, N=4096, K=1024)
    k_gemm<<<dim3(XROW / 64, MROWS / 128), 256, 0, stream>>>(
        hn_h, hn_l, ipw_h, ipw_l, xz, XROW, DM, DM, DM, XROW, 0, nullptr,
        nullptr, nullptr);

    // 3) x = silu(depthwise_conv(x) + b)  (-> f32 + bf16 hi/lo)
    k_conv_silu<<<(BSZ * LSEQ * DI) / 256, 256, 0, stream>>>(
        xz, conv_w, conv_b, xc, xch, xcl);

    // 4) x_dbl = x @ x_proj_w^T  (M=2048, N=96, K=2048) -> f32 + bf16 hi/lo
    k_gemm<<<dim3((XDROW + 63) / 64, MROWS / 128), 256, 0, stream>>>(
        xch, xcl, xpw_h, xpw_l, xdbl, XDROW, DI, DI, DI, XDROW, 0, nullptr,
        xdbl_h, xdbl_l);

    // 5) dt = softplus(x_dbl[:, :64] @ dt_proj_w^T + dt_proj_b)
    //    (M=2048, N=2048, K=64, lda=96)
    k_gemm<<<dim3(DI / 64, MROWS / 128), 256, 0, stream>>>(
        xdbl_h, xdbl_l, dtw_h, dtw_l, dtb, DI, DTR, XDROW, DTR, DI, 1,
        dt_proj_b, nullptr, nullptr);

    // 6) selective scan + D skip + silu(z) gate (-> bf16 hi/lo)
    k_scan<<<(BSZ * DI) / 256, 256, 0, stream>>>(
        dtb, xc, xdbl, xz, A_log, Dw, yh, yl);

    // 7) out = y @ out_proj_w^T  (M=2048, N=1024, K=2048)
    k_gemm<<<dim3(DM / 64, MROWS / 128), 256, 0, stream>>>(
        yh, yl, opw_h, opw_l, out_main, DM, DI, DI, DI, DM, 0, nullptr,
        nullptr, nullptr);
}
